// ShuffleFeature1_40570261078026
// MI455X (gfx1250) — matrix operand order repaired and checked by reading, not compiled, not run on hardware
//
#include <hip/hip_runtime.h>
#include <hip/hip_bf16.h>

// bf16 storage + WMMA vector types
typedef __bf16 bf16_t;
typedef __attribute__((ext_vector_type(16))) __bf16 v16bf;
typedef __attribute__((ext_vector_type(8)))  __bf16 v8bf;
typedef __attribute__((ext_vector_type(8)))  float  v8f;

__device__ __forceinline__ float mishf(float x) {
  float sp = (x > 20.f) ? x : log1pf(__expf(x));   // softplus, overflow-safe
  return x * tanhf(sp);
}
__device__ __forceinline__ float sigm(float x) { return 1.f / (1.f + __expf(-x)); }

// ---------------------------------------------------------------------------
// Kernel 1: f = mish(conv3x3_{1->4}(x)); 128 images of 256x256, bf16 output.
// ---------------------------------------------------------------------------
__global__ void conv1_mish_k(const float* __restrict__ x,
                             const float* __restrict__ W1,
                             const float* __restrict__ b1,
                             bf16_t* __restrict__ f) {
  int idx = blockIdx.x * blockDim.x + threadIdx.x;   // 0 .. 128*65536-1
  int img = idx >> 16;
  int rem = idx & 65535;
  int h = rem >> 8, w = rem & 255;
  const float* xin = x + ((size_t)img << 16);
  float v[9];
#pragma unroll
  for (int dy = 0; dy < 3; ++dy)
#pragma unroll
    for (int dx = 0; dx < 3; ++dx) {
      int hh = h + dy - 1, ww = w + dx - 1;
      v[dy * 3 + dx] = (hh >= 0 && hh < 256 && ww >= 0 && ww < 256)
                           ? xin[hh * 256 + ww] : 0.f;
    }
#pragma unroll
  for (int ch = 0; ch < 4; ++ch) {
    float acc = b1[ch];
#pragma unroll
    for (int t = 0; t < 9; ++t) acc += W1[ch * 9 + t] * v[t];
    f[(((size_t)img * 4 + ch) << 16) + rem] = (bf16_t)mishf(acc);
  }
}

// ---------------------------------------------------------------------------
// Kernel 2: fused {1x1 red -> relu -> 1x1 span -> per-pixel 3x3 on top half}
// plus channel-rolled bottom-half copy -> `shuffled` (B,64,256,256) bf16.
// ---------------------------------------------------------------------------
__global__ void shuffle_build_k(const bf16_t* __restrict__ f,
                                const float* __restrict__ Wred,
                                const float* __restrict__ bred,
                                const float* __restrict__ Wspan,
                                const float* __restrict__ bspan,
                                bf16_t* __restrict__ sh) {
  int idx = blockIdx.x * blockDim.x + threadIdx.x;   // 0 .. 128*128*256-1
  int img = idx >> 15;
  int rem = idx & 32767;
  int h2 = rem >> 8, w = rem & 255;

  float bot[4];
  float t = bred[0];
#pragma unroll
  for (int ch = 0; ch < 4; ++ch) {
    bot[ch] = (float)f[(((size_t)img * 4 + ch) << 16) + ((128 + h2) << 8) + w];
    t += Wred[ch] * bot[ch];
  }
  t = fmaxf(t, 0.f);
  float ker[9];
#pragma unroll
  for (int k = 0; k < 9; ++k) ker[k] = Wspan[k] * t + bspan[k];

  int b = img >> 4, c = img & 15;
#pragma unroll
  for (int ch = 0; ch < 4; ++ch) {
    const bf16_t* fch = f + (((size_t)img * 4 + ch) << 16);
    float acc = 0.f;
#pragma unroll
    for (int dy = 0; dy < 3; ++dy) {
      int hh = h2 + dy - 1;
      if (hh < 0 || hh >= 128) continue;     // top-half tensor is padded, not f
#pragma unroll
      for (int dx = 0; dx < 3; ++dx) {
        int ww = w + dx - 1;
        if (ww < 0 || ww >= 256) continue;
        acc += ker[dy * 3 + dx] * (float)fch[(hh << 8) + ww];
      }
    }
    sh[(((size_t)(b * 64 + c * 4 + ch)) << 16) + (h2 << 8) + w] = (bf16_t)acc;
    int cPrev = (c + 15) & 15;   // roll(bo,-1,axis=1)
    sh[(((size_t)(b * 64 + cPrev * 4 + ch)) << 16) + ((128 + h2) << 8) + w] =
        (bf16_t)bot[ch];
  }
}

// ---------------------------------------------------------------------------
// Kernel 3: conv2 (64->16, 3x3) implicit GEMM on v_wmma_f32_16x16x32_bf16.
// K-order chosen as k = tap*64 + cin so each K=32 step has a compile-time
// (dy,dx) and contiguous cin -> A/B fragments are 2x ds_load_b128 each.
// Fused mish + global-mean-pool epilogue. Block = 128 thr (4 waves),
// tile = 64 pixels x 16 out channels, 18 WMMA per wave.
// ---------------------------------------------------------------------------
#define XLW  66   // 64-pixel tile + 2 halo columns
#define CINP 72   // cin padded 64->72 (36-dword lane stride: conflict-free b128)
#define WROW 584  // 576 k padded ->584   (292-dword lane stride: conflict-free)
__global__ void conv2_wmma_k(const bf16_t* __restrict__ sh,
                             const float* __restrict__ W2,
                             const float* __restrict__ b2,
                             float* __restrict__ pooled) {
  __shared__ __align__(16) bf16_t xlds[3 * XLW * CINP];  // [row][col][cin]
  __shared__ __align__(16) bf16_t wlds[16 * WROW];       // [co][k=tap*64+cin]

  int tid = threadIdx.x;                 // 0..127
  int blk = blockIdx.x;                  // 8*256*4 blocks
  int tile = blk & 3;
  int h = (blk >> 2) & 255;
  int b = blk >> 10;
  int w0 = tile * 64;

  // stage weights: W2[co][cin][dy][dx] -> wlds[co][tap*64+cin]
  for (int co = 0; co < 16; ++co)
    for (int k = tid; k < 576; k += 128) {
      int tap = k >> 6, cin = k & 63;
      wlds[co * WROW + k] = (bf16_t)W2[(size_t)co * 576 + cin * 9 + tap];
    }
  // stage x tile, cin innermost (col-consecutive threads => coalesced reads)
  for (int e = tid; e < 3 * 64 * XLW; e += 128) {
    int col = e % XLW;
    int t2 = e / XLW;
    int cin = t2 & 63;
    int r = t2 >> 6;
    int gh = h + r - 1, gw = w0 + col - 1;
    float val = 0.f;
    if (gh >= 0 && gh < 256 && gw >= 0 && gw < 256)
      val = (float)sh[(((size_t)(b * 64 + cin)) << 16) + (gh << 8) + gw];
    xlds[(r * XLW + col) * CINP + cin] = (bf16_t)val;
  }
  __syncthreads();

  int lane = tid & 31;
  int wave = tid >> 5;
  int half = lane >> 4;      // 0: lanes 0-15, 1: lanes 16-31
  int mn = lane & 15;        // M (pixel) for A/D, N (out channel) for B/D
  int lw = wave * 16 + mn;   // pixel column within 64-wide tile

  // per-lane base pointers; all per-step offsets are compile-time immediates
  const bf16_t* xbase = &xlds[lw * CINP + half * 8];
  const bf16_t* wbase = &wlds[mn * WROW + half * 8];

  v8f acc = {0.f, 0.f, 0.f, 0.f, 0.f, 0.f, 0.f, 0.f};
#pragma unroll
  for (int kb = 0; kb < 18; ++kb) {
    const int tap = kb >> 1;                 // compile-time after unroll
    const int dy = tap / 3, dx = tap - dy * 3;
    const int xoff = (dy * XLW + dx) * CINP + (kb & 1) * 32;
    // A fragment: elements 0..7 = cin0..cin0+7, 8..15 = cin0+16..cin0+23
    v8bf alo = *(const v8bf*)(xbase + xoff);
    v8bf ahi = *(const v8bf*)(xbase + xoff + 16);
    v16bf a = __builtin_shufflevector(alo, ahi, 0, 1, 2, 3, 4, 5, 6, 7,
                                      8, 9, 10, 11, 12, 13, 14, 15);
    // B fragment: elements 0..7 = k0..k0+7, 8..15 = k0+16..k0+23 (k0 = kb*32 + half*8)
    v8bf blo = *(const v8bf*)(wbase + kb * 32);
    v8bf bhi = *(const v8bf*)(wbase + kb * 32 + 16);
    v16bf bm = __builtin_shufflevector(blo, bhi, 0, 1, 2, 3, 4, 5, 6, 7,
                                       8, 9, 10, 11, 12, 13, 14, 15);
    acc = __builtin_amdgcn_wmma_f32_16x16x32_bf16(
        /*neg_a=*/false, a, /*neg_b=*/false, bm,
        /*c_mod=*/(short)0, acc, /*reuse_a=*/false, /*reuse_b=*/false);
  }

  // fused epilogue: mish + mean-pool partial. D layout: VGPR r -> M=r (lanes
  // 0-15) / M=8+r (lanes 16-31), N = lane&15. Sum rows, fold halves, atomic.
  float bias = b2[mn];
  float s = 0.f;
#pragma unroll
  for (int r = 0; r < 8; ++r) s += mishf(acc[r] + bias);
  s += __shfl_down(s, 16, 32);
  if (lane < 16) atomicAdd(&pooled[b * 16 + mn], s);
}

// ---------------------------------------------------------------------------
// Small kernels: pooled zero-init, ECA gate, final scale.
// ---------------------------------------------------------------------------
__global__ void zero_pooled_k(float* __restrict__ pooled) {
  pooled[threadIdx.x] = 0.f;   // 128 threads
}

__global__ void gate_k(const float* __restrict__ pooled,
                       const float* __restrict__ Weca,
                       float* __restrict__ gate) {
  __shared__ float sm[128];
  int tid = threadIdx.x;       // 128 = 8 batches x 16 channels
  float m = pooled[tid] * (1.f / 65536.f);   // mean over H*W
  sm[tid] = m;
  __syncthreads();
  int c = tid & 15;
  float left  = (c > 0)  ? sm[tid - 1] : 0.f;
  float right = (c < 15) ? sm[tid + 1] : 0.f;
  gate[tid] = sigm(Weca[0] * left + Weca[1] * m + Weca[2] * right);
}

__global__ void scale_k(const float4* __restrict__ x4,
                        const float* __restrict__ gate,
                        float4* __restrict__ out4) {
  int idx = blockIdx.x * blockDim.x + threadIdx.x;  // 0 .. 2097151
  float g = gate[idx >> 14];                        // 16384 float4 per channel
  float4 v = x4[idx];
  v.x *= g; v.y *= g; v.z *= g; v.w *= g;
  out4[idx] = v;
}

// ---------------------------------------------------------------------------
// Launch. Workspace: f (64 MiB bf16) | shuffled (64 MiB bf16) | pooled | gate.
// ---------------------------------------------------------------------------
extern "C" void kernel_launch(void* const* d_in, const int* in_sizes, int n_in,
                              void* d_out, int out_size, void* d_ws, size_t ws_size,
                              hipStream_t stream) {
  const float* x     = (const float*)d_in[0];
  const float* W1    = (const float*)d_in[1];
  const float* b1    = (const float*)d_in[2];
  const float* Wred  = (const float*)d_in[3];
  const float* bred  = (const float*)d_in[4];
  const float* Wspan = (const float*)d_in[5];
  const float* bspan = (const float*)d_in[6];
  const float* W2    = (const float*)d_in[7];
  const float* b2    = (const float*)d_in[8];
  const float* Weca  = (const float*)d_in[9];

  char* ws = (char*)d_ws;
  bf16_t* f      = (bf16_t*)ws;                       // 128*4*65536 bf16 = 64 MiB
  bf16_t* shuf   = (bf16_t*)(ws + (size_t)67108864);  // 8*64*65536 bf16 = 64 MiB
  float*  pooled = (float*)(ws + (size_t)134217728);  // 128 floats
  float*  gate   = pooled + 128;                      // 128 floats

  conv1_mish_k<<<32768, 256, 0, stream>>>(x, W1, b1, f);
  shuffle_build_k<<<16384, 256, 0, stream>>>(f, Wred, bred, Wspan, bspan, shuf);
  zero_pooled_k<<<1, 128, 0, stream>>>(pooled);
  conv2_wmma_k<<<8192, 128, 0, stream>>>(shuf, W2, b2, pooled);
  gate_k<<<1, 128, 0, stream>>>(pooled, Weca, gate);
  scale_k<<<8192, 256, 0, stream>>>((const float4*)x, gate, (float4*)d_out);
}
